// MultiHeadAttention_36928128811505
// MI455X (gfx1250) — matrix operand order repaired and checked by reading, not compiled, not run on hardware
//
#include <hip/hip_runtime.h>
#include <stdint.h>

#define B_   32
#define D_   512
#define S_   512
#define H_   8
#define DH_  64
#define NEGINF (-1000000000.0f)
#define QS   72   // padded LDS row stride (bf16 elements) for 64-wide tiles

typedef __attribute__((ext_vector_type(16))) __bf16 v16bf;
typedef __attribute__((ext_vector_type(8)))  __bf16 v8bf;
typedef __attribute__((ext_vector_type(4)))  __bf16 v4bf;
typedef __attribute__((ext_vector_type(8)))  float  v8f;

__device__ __forceinline__ v16bf cat8(v8bf a, v8bf b) {
  return __builtin_shufflevector(a, b, 0,1,2,3,4,5,6,7,8,9,10,11,12,13,14,15);
}

// -------- Kernel 0: W f32 -> bf16 ---------------------------------------
__global__ __launch_bounds__(256) void convw_kernel(const float* __restrict__ W,
                                                    __bf16* __restrict__ Wbf) {
  size_t i = ((size_t)blockIdx.x * blockDim.x + threadIdx.x) * 4;
  float4 f = *(const float4*)(W + i);
  v4bf o;
  o[0] = (__bf16)f.x; o[1] = (__bf16)f.y; o[2] = (__bf16)f.z; o[3] = (__bf16)f.w;
  *(v4bf*)(Wbf + i) = o;
}

// -------- Kernel 1: flash attention, writes x = (q+attn - m_v)/s_v -------
__global__ __launch_bounds__(128) void attn_kernel(
    const float* __restrict__ q, const float* __restrict__ k,
    const float* __restrict__ v, const unsigned char* __restrict__ mask,
    const float* __restrict__ gamma, const float* __restrict__ beta,
    const float* __restrict__ mqk, const float* __restrict__ sqk,
    const float* __restrict__ mv, const float* __restrict__ sv,
    float* __restrict__ xout)
{
  __shared__ __bf16 Qs[64 * QS];        // [query_local][d]
  __shared__ __bf16 Ks[64 * QS];        // [d][kcol]
  __shared__ __bf16 Vts[64 * QS];       // [kcol][d]   (transposed V)
  __shared__ __bf16 Ps[4 * 16 * QS];    // per-wave P strip [16][64]

  const int tid   = threadIdx.x;
  const int wave  = tid >> 5;
  const int lane  = tid & 31;
  const int lhalf = lane & 15;
  const int hi    = lane >> 4;

  const int bx = blockIdx.x;
  const int qt = bx & 7;
  const int h  = (bx >> 3) & 7;
  const int b  = bx >> 6;
  const int q0 = qt * 64;

  const float* qb = q + ((size_t)b * D_ + h * DH_) * S_;
  const float* kb = k + ((size_t)b * D_ + h * DH_) * S_;
  const float* vb = v + ((size_t)b * D_ + h * DH_) * S_;
  const unsigned char* mb = mask + (size_t)b * S_;

  // fold per-head affine + 1/sqrt(dh) into alpha*s + betaE
  const float g = gamma[h], be = beta[h], mm = mqk[h], ss = sqk[h];
  const float alpha = g / (ss * 8.0f);          // sqrt(64) == 8
  const float betaE = be - mm * g / ss;

  const int rr  = tid >> 4;          // 0..7
  const int cc  = (tid & 15) << 2;   // 0,4,..,60

  // ---- stage Q tile (transposed: Qs[s_local][d]) ----
  #pragma unroll
  for (int pass = 0; pass < 8; ++pass) {
    int dl = pass * 8 + rr;
    float4 f = *(const float4*)(qb + (size_t)dl * S_ + q0 + cc);
    Qs[(cc + 0) * QS + dl] = (__bf16)f.x;
    Qs[(cc + 1) * QS + dl] = (__bf16)f.y;
    Qs[(cc + 2) * QS + dl] = (__bf16)f.z;
    Qs[(cc + 3) * QS + dl] = (__bf16)f.w;
  }
  __syncthreads();

  // ---- hoist Q A-fragments (ISA 7.12.2 16-bit A layout) ----
  v16bf aQ[2];
  {
    const int arow = wave * 16 + lhalf;
    #pragma unroll
    for (int kc = 0; kc < 2; ++kc) {
      int e0 = 32 * kc + (hi ? 8 : 0);
      v8bf a0 = *(const v8bf*)(&Qs[arow * QS + e0]);
      v8bf a1 = *(const v8bf*)(&Qs[arow * QS + e0 + 16]);
      aQ[kc] = cat8(a0, a1);
    }
  }

  float m_run[8], l_run[8];
  v8f acc[4] = {};                     // [dh tile][row]  f32 accumulators
  #pragma unroll
  for (int r = 0; r < 8; ++r) { m_run[r] = -3.0e38f; l_run[r] = 0.0f; }

  __bf16* Pw = &Ps[wave * 16 * QS];

  for (int kt = 0; kt < 8; ++kt) {
    const int kb0 = kt * 64;
    // ---- stage K (natural) and V (transposed) tiles ----
    #pragma unroll
    for (int pass = 0; pass < 8; ++pass) {
      int dl = pass * 8 + rr;
      float4 fk = *(const float4*)(kb + (size_t)dl * S_ + kb0 + cc);
      v4bf kk;
      kk[0] = (__bf16)fk.x; kk[1] = (__bf16)fk.y;
      kk[2] = (__bf16)fk.z; kk[3] = (__bf16)fk.w;
      *(v4bf*)(&Ks[dl * QS + cc]) = kk;
      float4 fv = *(const float4*)(vb + (size_t)dl * S_ + kb0 + cc);
      Vts[(cc + 0) * QS + dl] = (__bf16)fv.x;
      Vts[(cc + 1) * QS + dl] = (__bf16)fv.y;
      Vts[(cc + 2) * QS + dl] = (__bf16)fv.z;
      Vts[(cc + 3) * QS + dl] = (__bf16)fv.w;
      if (kt < 7) {   // pull next tile toward the caches (global_prefetch_b8)
        __builtin_prefetch(kb + (size_t)dl * S_ + kb0 + 64 + cc, 0, 1);
        __builtin_prefetch(vb + (size_t)dl * S_ + kb0 + 64 + cc, 0, 1);
      }
    }
    __syncthreads();

    // ---- scores: S = Q^T K  (WMMA bf16, f32 acc) + affine + mask ----
    float sc[4][8];
    bool  mcol[4];
    #pragma unroll
    for (int nt = 0; nt < 4; ++nt) {
      v8f s = {};
      #pragma unroll
      for (int kc = 0; kc < 2; ++kc) {
        v16bf bK;
        #pragma unroll
        for (int s = 0; s < 16; ++s)
          bK[s] = Ks[(32 * kc + 8 * hi + (s & 7) + 16 * (s >> 3)) * QS + 16 * nt + lhalf];
        s = __builtin_amdgcn_wmma_f32_16x16x32_bf16(false, aQ[kc], false, bK,
                                                    (short)0, s, false, false);
      }
      mcol[nt] = (mb[kb0 + 16 * nt + lhalf] != 0);
      #pragma unroll
      for (int r = 0; r < 8; ++r)
        sc[nt][r] = mcol[nt] ? NEGINF : (s[r] * alpha + betaE);
    }

    // ---- online softmax (row-wise across 16-lane halves) ----
    float pv[4][8];
    #pragma unroll
    for (int r = 0; r < 8; ++r) {
      float mx = fmaxf(fmaxf(sc[0][r], sc[1][r]), fmaxf(sc[2][r], sc[3][r]));
      #pragma unroll
      for (int off = 1; off < 16; off <<= 1)
        mx = fmaxf(mx, __shfl_xor(mx, off, 32));
      float mn    = fmaxf(m_run[r], mx);
      float scale = __expf(m_run[r] - mn);
      float rs = 0.0f;
      #pragma unroll
      for (int nt = 0; nt < 4; ++nt) {
        float p = mcol[nt] ? 0.0f : __expf(sc[nt][r] - mn);
        pv[nt][r] = p;
        rs += p;
      }
      #pragma unroll
      for (int off = 1; off < 16; off <<= 1)
        rs += __shfl_xor(rs, off, 32);
      l_run[r] = l_run[r] * scale + rs;
      m_run[r] = mn;
      #pragma unroll
      for (int t = 0; t < 4; ++t) acc[t][r] *= scale;
    }

    // ---- bounce P through per-wave LDS strip (LDS is in-order per wave) ----
    #pragma unroll
    for (int nt = 0; nt < 4; ++nt)
      #pragma unroll
      for (int r = 0; r < 8; ++r)
        Pw[(r + 8 * hi) * QS + 16 * nt + lhalf] = (__bf16)pv[nt][r];

    // ---- O += P * V^T ----
    #pragma unroll
    for (int dt = 0; dt < 4; ++dt) {
      #pragma unroll
      for (int kc = 0; kc < 2; ++kc) {
        int e0 = 32 * kc + (hi ? 8 : 0);
        v8bf p0 = *(const v8bf*)(&Pw[lhalf * QS + e0]);
        v8bf p1 = *(const v8bf*)(&Pw[lhalf * QS + e0 + 16]);
        v16bf aP = cat8(p0, p1);
        v16bf bV;
        #pragma unroll
        for (int s = 0; s < 16; ++s)
          bV[s] = Vts[(16 * dt + lhalf) * QS + 32 * kc + 8 * hi + (s & 7) + 16 * (s >> 3)];
        acc[dt] = __builtin_amdgcn_wmma_f32_16x16x32_bf16(false, aP, false, bV,
                                                          (short)0, acc[dt], false, false);
      }
    }
    __syncthreads();
  }

  // ---- epilogue: x = (q + attn/l - m_v) / s_v ----
  float inv[8];
  #pragma unroll
  for (int r = 0; r < 8; ++r) inv[r] = (l_run[r] > 0.0f) ? (1.0f / l_run[r]) : 0.0f;
  #pragma unroll
  for (int dt = 0; dt < 4; ++dt) {
    int dg = h * DH_ + 16 * dt + lhalf;
    float mvv = mv[dg];
    float isv = 1.0f / sv[dg];
    #pragma unroll
    for (int r = 0; r < 8; ++r) {
      int sg = q0 + wave * 16 + 8 * hi + r;
      size_t idx = ((size_t)b * D_ + dg) * S_ + sg;
      float xv = q[idx] + acc[dt][r] * inv[r];
      xout[idx] = (xv - mvv) * isv;
    }
  }
}

// -------- Kernel 2: out = silu(W x + b)  (bf16 WMMA GEMM) ----------------
// W tile (already bf16) is DMA'd global->LDS with GLOBAL_LOAD_ASYNC_TO_LDS
// (ASYNCcnt path, no VGPR staging); X tile converts f32->bf16 through VALU.
__global__ __launch_bounds__(128) void proj_kernel(
    const __bf16* __restrict__ Wbf, const float* __restrict__ X,
    const float* __restrict__ bias, float* __restrict__ out)
{
  __shared__ __bf16 Ws[64 * QS];   // [o_local][d]
  __shared__ __bf16 Xs[64 * QS];   // [d][s]

  const int tid   = threadIdx.x;
  const int wave  = tid >> 5;
  const int lane  = tid & 31;
  const int lhalf = lane & 15;
  const int hi    = lane >> 4;

  const int bx = blockIdx.x;
  const int st = bx & 7;
  const int ot = (bx >> 3) & 7;
  const int b  = bx >> 6;
  const int s0 = st * 64, o0 = ot * 64;

  const int rr = tid >> 4;
  const int cc = (tid & 15) << 2;

  v8f acc[4] = {};

  for (int kt = 0; kt < 8; ++kt) {
    const int k0 = kt * 64;

    // ---- async DMA W tile into LDS: 512 x 16B chunks, 4 per thread ----
    #pragma unroll
    for (int i = 0; i < 4; ++i) {
      int c   = tid + 128 * i;            // 0..511
      int row = c >> 3;                   // 0..63
      int col = (c & 7) * 8;              // 0,8,..,56 (bf16 elems)
      unsigned lds_off = (unsigned)(uintptr_t)(&Ws[row * QS + col]);
      unsigned goff = (((unsigned)(o0 + row)) * (unsigned)D_ +
                       (unsigned)(k0 + col)) * 2u;   // byte offset
      asm volatile("global_load_async_to_lds_b128 %0, %1, %2"
                   :: "v"(lds_off), "v"(goff), "s"(Wbf)
                   : "memory");
    }

    // ---- stage X tile (f32 -> bf16 conversion needs VALU anyway) ----
    #pragma unroll
    for (int pass = 0; pass < 8; ++pass) {
      int row = pass * 8 + rr;
      float4 fx = *(const float4*)(X + ((size_t)b * D_ + k0 + row) * S_ + s0 + cc);
      v4bf xb;
      xb[0] = (__bf16)fx.x; xb[1] = (__bf16)fx.y;
      xb[2] = (__bf16)fx.z; xb[3] = (__bf16)fx.w;
      *(v4bf*)(&Xs[row * QS + cc]) = xb;
    }

    asm volatile("s_wait_asynccnt 0x0" ::: "memory");  // this wave's DMAs done
    __syncthreads();                                   // all waves' tiles done

    v16bf aW[2];
    {
      const int arow = wave * 16 + lhalf;
      #pragma unroll
      for (int kc = 0; kc < 2; ++kc) {
        int e0 = 32 * kc + (hi ? 8 : 0);
        v8bf a0 = *(const v8bf*)(&Ws[arow * QS + e0]);
        v8bf a1 = *(const v8bf*)(&Ws[arow * QS + e0 + 16]);
        aW[kc] = cat8(a0, a1);
      }
    }
    #pragma unroll
    for (int nt = 0; nt < 4; ++nt) {
      #pragma unroll
      for (int kc = 0; kc < 2; ++kc) {
        v16bf bX;
        #pragma unroll
        for (int s = 0; s < 16; ++s)
          bX[s] = Xs[(32 * kc + 8 * hi + (s & 7) + 16 * (s >> 3)) * QS + 16 * nt + lhalf];
        acc[nt] = __builtin_amdgcn_wmma_f32_16x16x32_bf16(false, aW[kc], false, bX,
                                                          (short)0, acc[nt], false, false);
      }
    }
    __syncthreads();
  }

  #pragma unroll
  for (int nt = 0; nt < 4; ++nt) {
    int sg = s0 + 16 * nt + lhalf;
    #pragma unroll
    for (int r = 0; r < 8; ++r) {
      int og = o0 + wave * 16 + 8 * hi + r;
      float xv = acc[nt][r] + bias[og];
      float y  = xv / (1.0f + __expf(-xv));   // silu
      out[((size_t)b * D_ + og) * S_ + sg] = y;
    }
  }
}

extern "C" void kernel_launch(void* const* d_in, const int* in_sizes, int n_in,
                              void* d_out, int out_size, void* d_ws, size_t ws_size,
                              hipStream_t stream) {
  const float*         q     = (const float*)d_in[0];
  const float*         k     = (const float*)d_in[1];
  const float*         v     = (const float*)d_in[2];
  const unsigned char* mask  = (const unsigned char*)d_in[3];
  const float*         gamma = (const float*)d_in[4];
  const float*         beta  = (const float*)d_in[5];
  const float*         mqk   = (const float*)d_in[6];
  const float*         sqk   = (const float*)d_in[7];
  const float*         mv    = (const float*)d_in[8];
  const float*         sv    = (const float*)d_in[9];
  const float*         W     = (const float*)d_in[10];
  const float*         bias  = (const float*)d_in[11];
  float*               out   = (float*)d_out;

  // workspace: x intermediate (B*D*S f32 = 32 MB), then bf16 W (512 KB)
  float*  Xws = (float*)d_ws;
  __bf16* Wbf = (__bf16*)((char*)d_ws + (size_t)B_ * D_ * S_ * sizeof(float));

  convw_kernel<<<(D_ * D_ / 4) / 256, 256, 0, stream>>>(W, Wbf);
  attn_kernel<<<B_ * H_ * (S_ / 64), 128, 0, stream>>>(
      q, k, v, mask, gamma, beta, mqk, sqk, mv, sv, Xws);
  proj_kernel<<<B_ * (D_ / 64) * (S_ / 64), 128, 0, stream>>>(Wbf, Xws, bias, out);
}
